// EnformerAttention_15556371546222
// MI455X (gfx1250) — compile-verified, matrix operand order checked by reading, not run on hardware
//
#include <hip/hip_runtime.h>
#include <hip/hip_bf16.h>
#include <math.h>

typedef __bf16 bf16;
typedef __attribute__((ext_vector_type(16))) __bf16 v16bf;
typedef __attribute__((ext_vector_type(8)))  float  v8f;
typedef __attribute__((ext_vector_type(4)))  unsigned int u32x4;
typedef __attribute__((ext_vector_type(8)))  int i32x8;
typedef __attribute__((ext_vector_type(4)))  int i32x4;

#define HEADS 8
#define DKEY  64
#define DVAL  192
#define DIM   1536
#define SEQ   1536
#define BATCH 4
#define NPOS  3072          /* 2*SEQ-1 = 3071, padded to 3072 */
#define QKDIM (HEADS*DKEY)  /* 512 */
#define MROWS (BATCH*SEQ)   /* 6144 */

/* LDS B-tile row pitch: 32 bf16 (64B) + TDM pad of 4 dwords (16B) = 80B = 40 elems.
   ds_load_b128 bank = 20*row mod 64 -> 16 distinct 4-bank groups: conflict-free. */
#define LDS_ROW 40

#if __has_builtin(__builtin_amdgcn_tensor_load_to_lds)
#define USE_TDM 1
#else
#define USE_TDM 0
#endif

// ---------------- WMMA fragment helpers (wave32, 16x16x32 bf16, f32 accum) ----------------
// 16-bit fragment layout (ISA 7.12.2): lane row r=lane&15, kbase=(lane>>4)*8,
// elems 0..7 -> K=kb..kb+7, elems 8..15 -> K=kb+16..kb+23  => two contiguous 16B loads.

__device__ __forceinline__ v16bf load_frag16x32(const bf16* __restrict__ p, int ld) {
    int lane = (int)(threadIdx.x & 31u);
    int r    = lane & 15;
    int kb   = (lane >> 4) << 3;
    const bf16* q = p + (size_t)r * ld + kb;
    v16bf f;
    *(uint4*)&f         = *(const uint4*)(q);
    *(((uint4*)&f) + 1) = *(const uint4*)(q + 16);
    return f;
}

__device__ __forceinline__ v16bf load_frag_lds(const bf16* p) {
    int lane = (int)(threadIdx.x & 31u);
    int r    = lane & 15;
    int kb   = (lane >> 4) << 3;
    const bf16* q = p + r * LDS_ROW + kb;
    v16bf f;
    *(uint4*)&f         = *(const uint4*)(q);
    *(((uint4*)&f) + 1) = *(const uint4*)(q + 16);
    return f;
}

__device__ __forceinline__ v8f wmma_bf16(v16bf a, v16bf b, v8f c) {
    return __builtin_amdgcn_wmma_f32_16x16x32_bf16(false, a, false, b, (short)0, c,
                                                   false, false);
}

// ---------------- Tensor Data Mover: async load of one 64x32 bf16 B-tile into LDS --------

__device__ __forceinline__ void tdm_load_b_tile(const bf16* gsrc, int ldb, unsigned lds_off) {
#if USE_TDM
    unsigned long long ga = (unsigned long long)(size_t)gsrc;
    u32x4 g0;
    g0[0] = 1u;                                                // count=1 (valid D#)
    g0[1] = lds_off;                                           // LDS byte address
    g0[2] = (unsigned)ga;                                      // global addr [31:0]
    g0[3] = (unsigned)((ga >> 32) & 0x01FFFFFFu) | (2u << 30); // addr [56:32] | type=2
    i32x8 g1;
    g1[0] = (int)((1u << 16)      /* data_size = 2 bytes            */
                | (1u << 20)      /* pad_enable                     */
                | (3u << 22)      /* pad_interval: 16 dwords (64B)  */
                | (3u << 25));    /* pad_amount  : 4 dwords  (16B)  */
    g1[1] = (int)(32u << 16);     // tensor_dim0 = 32 (lo16 @ bits 63:48)
    g1[2] = (int)(64u << 16);     // tensor_dim0 hi=0 | tensor_dim1 = 64 (lo16)
    g1[3] = (int)(32u << 16);     // tensor_dim1 hi=0 | tile_dim0 = 32
    g1[4] = (int)(64u);           // tile_dim1 = 64 | tile_dim2 = 0
    g1[5] = (int)(unsigned)ldb;   // tensor_dim0_stride lo32 (elements)
    g1[6] = 0;                    // stride hi16 | tensor_dim1_stride lo16
    g1[7] = 0;
    i32x4 z4 = {0, 0, 0, 0};
    i32x8 z8 = {0, 0, 0, 0, 0, 0, 0, 0};
    __builtin_amdgcn_tensor_load_to_lds(g0, g1, z4, z4, z8, 0);
#else
    (void)gsrc; (void)ldb; (void)lds_off;
#endif
}

// ---------------- GEMM core: block = 4 waves -> 128x64 tile, B staged in LDS ------------
// Each wave: 32x64 (2 A-frags x 4 B-frags x 8 WMMA per 32-wide K step).

__device__ __forceinline__ void gemm_compute_chunk(const bf16* __restrict__ A, int lda, int k,
                                                   const bf16* sbuf, v8f (&acc)[2][4]) {
    v16bf af[2], bfr[4];
#pragma unroll
    for (int m = 0; m < 2; ++m)
        af[m] = load_frag16x32(A + (size_t)m * 16 * lda + k, lda);
#pragma unroll
    for (int n = 0; n < 4; ++n)
        bfr[n] = load_frag_lds(sbuf + (n * 16) * LDS_ROW);
#pragma unroll
    for (int m = 0; m < 2; ++m)
#pragma unroll
        for (int n = 0; n < 4; ++n)
            acc[m][n] = wmma_bf16(af[m], bfr[n], acc[m][n]);
}

__device__ __forceinline__ void gemm_tile_lds(const bf16* __restrict__ A, int lda,
                                              const bf16* __restrict__ BtTile, int ldb,
                                              int K, v8f (&acc)[2][4],
                                              bf16 (&sB)[2][64 * LDS_ROW]) {
    const int wave = (int)threadIdx.x >> 5;
    const int nk   = K >> 5;
#if USE_TDM
    if (wave == 0)
        tdm_load_b_tile(BtTile, ldb, (unsigned)(size_t)&sB[0][0]);
    for (int kc = 0; kc < nk; ++kc) {
        if (wave == 0) {
            if (kc + 1 < nk) {
                tdm_load_b_tile(BtTile + ((kc + 1) << 5), ldb,
                                (unsigned)(size_t)&sB[(kc + 1) & 1][0]);
                __builtin_amdgcn_s_wait_tensorcnt(1);  // chunk kc landed
            } else {
                __builtin_amdgcn_s_wait_tensorcnt(0);
            }
        }
        __syncthreads();                               // B chunk visible to all waves
        gemm_compute_chunk(A, lda, kc << 5, &sB[kc & 1][0], acc);
        __syncthreads();                               // done reading before reuse
    }
#else
    const int t = (int)threadIdx.x;
    for (int kc = 0; kc < nk; ++kc) {
#pragma unroll
        for (int i = 0; i < 2; ++i) {                  // 256 8-elem chunks / 128 threads
            int c  = t + i * 128;
            int y  = c >> 2;
            int xc = (c & 3) << 3;
            *(uint4*)&sB[0][y * LDS_ROW + xc] =
                *(const uint4*)(BtTile + (size_t)y * ldb + (kc << 5) + xc);
        }
        __syncthreads();
        gemm_compute_chunk(A, lda, kc << 5, &sB[0][0], acc);
        __syncthreads();
    }
#endif
}

// Block tile coords; grids are exact so no guard (keeps __syncthreads uniform).
__device__ __forceinline__ void block_tile(int M, int& tm, int& tn) {
    int tiles_m = M / 128;
    tm = (int)blockIdx.x % tiles_m;
    tn = (int)blockIdx.x / tiles_m;
}

// C/D f32 16x16 layout: lane col = lane&15, elem r -> row r + 8*(lane>>4).

// ---------------- conversion / staging kernels ----------------

__global__ void convert_bf16_kernel(const float* __restrict__ in, bf16* __restrict__ out, int n) {
    int i = (int)(blockIdx.x * blockDim.x + threadIdx.x);
    if (i < n) out[i] = (bf16)in[i];
}

__global__ void transpose_bf16_kernel(const float* __restrict__ in, bf16* __restrict__ out,
                                      int K, int N) {
    int t = (int)(blockIdx.x * blockDim.x + threadIdx.x);
    if (t >= K * N) return;
    int n = t / K;
    int k = t - n * K;
    out[t] = (bf16)in[(size_t)k * N + n];
}

__global__ void pos_embed_kernel(bf16* __restrict__ pos) {
    int r = (int)(blockIdx.x * blockDim.x + threadIdx.x);
    if (r >= NPOS) return;
    if (r == NPOS - 1) {
        for (int c = 0; c < 192; ++c) pos[(size_t)r * 192 + c] = (bf16)0.0f;
        return;
    }
    float d  = (float)r - 1535.0f;
    float ad = fabsf(d);
    float sg = (d > 0.0f) ? 1.0f : ((d < 0.0f) ? -1.0f : 0.0f);
    const float ln2 = 0.69314718055994531f;
    const float max_range = log2f((float)SEQ);
    float half[96];
    for (int j = 0; j < 32; ++j) {                       // exponential basis
        float e  = 3.0f + (max_range - 3.0f) * (float)j / 31.0f;
        half[j]  = expf(-ln2 / exp2f(e) * ad);
    }
    for (int j = 0; j < 32; ++j) {                       // central mask basis
        float cw = exp2f((float)(j + 1)) - 1.0f;
        half[32 + j] = (cw > ad) ? 1.0f : 0.0f;
    }
    float probs[32];                                     // gamma basis
    float pm = 0.0f;
    for (int j = 0; j < 32; ++j) {
        float mean = 48.0f * (float)(j + 1);
        float conc = (mean / 24.0f) * (mean / 24.0f);
        float rate = mean / 576.0f;
        float lu   = (ad > 0.0f) ? (conc - 1.0f) * logf(ad) : -__builtin_inff();
        lu -= rate * ad;
        float ln_  = lgammaf(conc) - conc * logf(rate);
        float p    = expf(lu - ln_) + 1e-8f;
        probs[j] = p;
        pm = fmaxf(pm, p);
    }
    for (int j = 0; j < 32; ++j) half[64 + j] = probs[j] / pm;
    for (int c = 0; c < 96; ++c) {
        pos[(size_t)r * 192 + c]      = (bf16)half[c];
        pos[(size_t)r * 192 + 96 + c] = (bf16)(sg * half[c]);
    }
}

// ---------------- GEMM kernels ----------------

__global__ void gemm_f32_kernel(const bf16* __restrict__ A, int lda,
                                const bf16* __restrict__ Bt, int ldb,
                                float* __restrict__ C, int ldc,
                                int M, int K) {
    __shared__ bf16 sB[2][64 * LDS_ROW];
    int tm, tn;
    block_tile(M, tm, tn);
    int wave = (int)threadIdx.x >> 5;
    v8f acc[2][4] = {};
    gemm_tile_lds(A + (size_t)(tm * 128 + wave * 32) * lda, lda,
                  Bt + (size_t)tn * 64 * ldb, ldb, K, acc, sB);
    int lane = (int)(threadIdx.x & 31u);
    int col0 = tn * 64 + (lane & 15);
    int r00  = tm * 128 + wave * 32 + ((lane >> 4) << 3);
#pragma unroll
    for (int m = 0; m < 2; ++m)
#pragma unroll
        for (int n = 0; n < 4; ++n) {
            int col = col0 + n * 16;
            int rb  = r00 + m * 16;
#pragma unroll
            for (int r = 0; r < 8; ++r)
                C[(size_t)(rb + r) * ldc + col] = acc[m][n][r];
        }
}

__global__ void gemm_bf16_kernel(const bf16* __restrict__ A, int lda,
                                 const bf16* __restrict__ Bt, int ldb,
                                 bf16* __restrict__ C, int ldc,
                                 int M, int K) {
    __shared__ bf16 sB[2][64 * LDS_ROW];
    int tm, tn;
    block_tile(M, tm, tn);
    int wave = (int)threadIdx.x >> 5;
    v8f acc[2][4] = {};
    gemm_tile_lds(A + (size_t)(tm * 128 + wave * 32) * lda, lda,
                  Bt + (size_t)tn * 64 * ldb, ldb, K, acc, sB);
    int lane = (int)(threadIdx.x & 31u);
    int col0 = tn * 64 + (lane & 15);
    int r00  = tm * 128 + wave * 32 + ((lane >> 4) << 3);
#pragma unroll
    for (int m = 0; m < 2; ++m)
#pragma unroll
        for (int n = 0; n < 4; ++n) {
            int col = col0 + n * 16;
            int rb  = r00 + m * 16;
#pragma unroll
            for (int r = 0; r < 8; ++r)
                C[(size_t)(rb + r) * ldc + col] = (bf16)acc[m][n][r];
        }
}

// q projection: scale by dk^-0.5, add both biases, emit two bf16 tensors
__global__ void gemm_q_kernel(const bf16* __restrict__ xb, const bf16* __restrict__ Wqt,
                              const float* __restrict__ biasC, const float* __restrict__ biasP,
                              bf16* __restrict__ qc, bf16* __restrict__ qp) {
    __shared__ bf16 sB[2][64 * LDS_ROW];
    int tm, tn;
    block_tile(MROWS, tm, tn);
    int wave = (int)threadIdx.x >> 5;
    v8f acc[2][4] = {};
    gemm_tile_lds(xb + (size_t)(tm * 128 + wave * 32) * DIM, DIM,
                  Wqt + (size_t)tn * 64 * DIM, DIM, DIM, acc, sB);
    const float scale = 0.125f;
    int lane = (int)(threadIdx.x & 31u);
    int col0 = tn * 64 + (lane & 15);
    int r00  = tm * 128 + wave * 32 + ((lane >> 4) << 3);
#pragma unroll
    for (int m = 0; m < 2; ++m)
#pragma unroll
        for (int n = 0; n < 4; ++n) {
            int col = col0 + n * 16;
            int rb  = r00 + m * 16;
            float bc = biasC[col], bp = biasP[col];
#pragma unroll
            for (int r = 0; r < 8; ++r) {
                float v = acc[m][n][r] * scale;
                size_t idx = (size_t)(rb + r) * QKDIM + col;
                qc[idx] = (bf16)(v + bc);
                qp[idx] = (bf16)(v + bp);
            }
        }
}

// v projection with transposed store: vt[(b*H+h)*DVAL + d][j]
__global__ void gemm_vt_kernel(const bf16* __restrict__ xb, const bf16* __restrict__ Wvt,
                               bf16* __restrict__ vt) {
    __shared__ bf16 sB[2][64 * LDS_ROW];
    int tm, tn;
    block_tile(MROWS, tm, tn);
    int wave = (int)threadIdx.x >> 5;
    v8f acc[2][4] = {};
    gemm_tile_lds(xb + (size_t)(tm * 128 + wave * 32) * DIM, DIM,
                  Wvt + (size_t)tn * 64 * DIM, DIM, DIM, acc, sB);
    int lane = (int)(threadIdx.x & 31u);
    int col0 = tn * 64 + (lane & 15);
    int r00  = tm * 128 + wave * 32 + ((lane >> 4) << 3);
#pragma unroll
    for (int m = 0; m < 2; ++m)
#pragma unroll
        for (int n = 0; n < 4; ++n) {
            int col = col0 + n * 16;
            int h = col / DVAL;
            int d = col - h * DVAL;
            int rb = r00 + m * 16;
#pragma unroll
            for (int r = 0; r < 8; ++r) {
                int mrow = rb + r;
                int b = mrow / SEQ;
                int j = mrow - b * SEQ;
                vt[(((size_t)b * HEADS + h) * DVAL + d) * SEQ + j] = (bf16)acc[m][n][r];
            }
        }
}

// rel_k = pos @ Wrel, stored per-head as (h, j', dk)
__global__ void gemm_relk_kernel(const bf16* __restrict__ pos, const bf16* __restrict__ Wrelt,
                                 bf16* __restrict__ relk) {
    __shared__ bf16 sB[2][64 * LDS_ROW];
    int tm, tn;
    block_tile(NPOS, tm, tn);
    int wave = (int)threadIdx.x >> 5;
    v8f acc[2][4] = {};
    gemm_tile_lds(pos + (size_t)(tm * 128 + wave * 32) * 192, 192,
                  Wrelt + (size_t)tn * 64 * 192, 192, 192, acc, sB);
    int lane = (int)(threadIdx.x & 31u);
    int col0 = tn * 64 + (lane & 15);
    int r00  = tm * 128 + wave * 32 + ((lane >> 4) << 3);
#pragma unroll
    for (int m = 0; m < 2; ++m)
#pragma unroll
        for (int n = 0; n < 4; ++n) {
            int col = col0 + n * 16;
            int h = col >> 6;
            int d = col & 63;
            int rb = r00 + m * 16;
#pragma unroll
            for (int r = 0; r < 8; ++r)
                relk[((size_t)h * NPOS + (rb + r)) * DKEY + d] = (bf16)acc[m][n][r];
        }
}

// final projection: out = ao @ Wout + b_out (f32 output)
__global__ void gemm_final_kernel(const bf16* __restrict__ ao, const bf16* __restrict__ Woutt,
                                  const float* __restrict__ bout, float* __restrict__ out) {
    __shared__ bf16 sB[2][64 * LDS_ROW];
    int tm, tn;
    block_tile(MROWS, tm, tn);
    int wave = (int)threadIdx.x >> 5;
    v8f acc[2][4] = {};
    gemm_tile_lds(ao + (size_t)(tm * 128 + wave * 32) * DIM, DIM,
                  Woutt + (size_t)tn * 64 * DIM, DIM, DIM, acc, sB);
    int lane = (int)(threadIdx.x & 31u);
    int col0 = tn * 64 + (lane & 15);
    int r00  = tm * 128 + wave * 32 + ((lane >> 4) << 3);
#pragma unroll
    for (int m = 0; m < 2; ++m)
#pragma unroll
        for (int n = 0; n < 4; ++n) {
            int col = col0 + n * 16;
            float bb = bout[col];
            int rb = r00 + m * 16;
#pragma unroll
            for (int r = 0; r < 8; ++r)
                out[(size_t)(rb + r) * DIM + col] = acc[m][n][r] + bb;
        }
}

// ---------------- fused relative-shift gather + softmax ----------------
// shifted_rel[i][j] = rel_unshift[i][SEQ-1 + j - i]; softmax over j; emit bf16 probs.
__global__ void softmax_kernel(const float* __restrict__ logits, const float* __restrict__ relbuf,
                               bf16* __restrict__ attn) {
    int i = (int)blockIdx.x;
    int t = (int)threadIdx.x;
    __shared__ float red[256];
    const float* lrow = logits + (size_t)i * SEQ;
    const float* rrow = relbuf + (size_t)i * NPOS + (SEQ - 1 - i);
    float vals[6];
    float mx = -3.4e38f;
#pragma unroll
    for (int c = 0; c < 6; ++c) {
        int j = t + c * 256;
        float v = lrow[j] + rrow[j];
        vals[c] = v;
        mx = fmaxf(mx, v);
    }
    red[t] = mx;
    __syncthreads();
    for (int s = 128; s > 0; s >>= 1) {
        if (t < s) red[t] = fmaxf(red[t], red[t + s]);
        __syncthreads();
    }
    mx = red[0];
    __syncthreads();
    float sum = 0.0f;
#pragma unroll
    for (int c = 0; c < 6; ++c) {
        vals[c] = expf(vals[c] - mx);
        sum += vals[c];
    }
    red[t] = sum;
    __syncthreads();
    for (int s = 128; s > 0; s >>= 1) {
        if (t < s) red[t] += red[t + s];
        __syncthreads();
    }
    float inv = 1.0f / red[0];
#pragma unroll
    for (int c = 0; c < 6; ++c)
        attn[(size_t)i * SEQ + t + c * 256] = (bf16)(vals[c] * inv);
}

// ---------------- host orchestration ----------------

extern "C" void kernel_launch(void* const* d_in, const int* in_sizes, int n_in,
                              void* d_out, int out_size, void* d_ws, size_t ws_size,
                              hipStream_t stream) {
    (void)in_sizes; (void)n_in; (void)out_size; (void)ws_size;
    const float* x     = (const float*)d_in[0];
    const float* Wq    = (const float*)d_in[1];
    const float* Wk    = (const float*)d_in[2];
    const float* Wv    = (const float*)d_in[3];
    const float* Wrel  = (const float*)d_in[4];
    const float* biasC = (const float*)d_in[5];
    const float* biasP = (const float*)d_in[6];
    const float* Wout  = (const float*)d_in[7];
    const float* bout  = (const float*)d_in[8];
    float* out = (float*)d_out;

    unsigned char* ws = (unsigned char*)d_ws;
    size_t off = 0;
    auto carve = [&](size_t bytes) -> void* {
        void* p = ws + off;
        off += (bytes + 255) & ~(size_t)255;
        return p;
    };
    bf16* xb    = (bf16*)carve((size_t)MROWS * DIM * 2);
    bf16* wqt   = (bf16*)carve((size_t)QKDIM * DIM * 2);
    bf16* wkt   = (bf16*)carve((size_t)QKDIM * DIM * 2);
    bf16* wvt   = (bf16*)carve((size_t)DIM * DIM * 2);
    bf16* wrelt = (bf16*)carve((size_t)QKDIM * 192 * 2);
    bf16* woutt = (bf16*)carve((size_t)DIM * DIM * 2);
    bf16* posb  = (bf16*)carve((size_t)NPOS * 192 * 2);
    bf16* qc    = (bf16*)carve((size_t)MROWS * QKDIM * 2);
    bf16* qp    = (bf16*)carve((size_t)MROWS * QKDIM * 2);
    bf16* kb    = (bf16*)carve((size_t)MROWS * QKDIM * 2);
    bf16* vt    = (bf16*)carve((size_t)BATCH * HEADS * DVAL * SEQ * 2);
    bf16* relk  = (bf16*)carve((size_t)HEADS * NPOS * DKEY * 2);
    bf16* ao    = (bf16*)carve((size_t)MROWS * DIM * 2);
    float* logits = (float*)carve((size_t)SEQ * SEQ * 4);
    float* relbuf = (float*)carve((size_t)SEQ * NPOS * 4);
    bf16* attn    = (bf16*)carve((size_t)SEQ * SEQ * 2);

    auto ceil_div = [](int a, int b) { return (a + b - 1) / b; };
    auto gemm_grid = [](int M, int N) { return (M / 128) * (N / 64); };

    // ---- staging ----
    convert_bf16_kernel<<<ceil_div(MROWS * DIM, 256), 256, 0, stream>>>(x, xb, MROWS * DIM);
    transpose_bf16_kernel<<<ceil_div(DIM * QKDIM, 256), 256, 0, stream>>>(Wq, wqt, DIM, QKDIM);
    transpose_bf16_kernel<<<ceil_div(DIM * QKDIM, 256), 256, 0, stream>>>(Wk, wkt, DIM, QKDIM);
    transpose_bf16_kernel<<<ceil_div(DIM * DIM, 256), 256, 0, stream>>>(Wv, wvt, DIM, DIM);
    transpose_bf16_kernel<<<ceil_div(192 * QKDIM, 256), 256, 0, stream>>>(Wrel, wrelt, 192, QKDIM);
    transpose_bf16_kernel<<<ceil_div(DIM * DIM, 256), 256, 0, stream>>>(Wout, woutt, DIM, DIM);
    pos_embed_kernel<<<ceil_div(NPOS, 256), 256, 0, stream>>>(posb);

    // ---- projections ----
    gemm_q_kernel<<<gemm_grid(MROWS, QKDIM), 128, 0, stream>>>(xb, wqt, biasC, biasP, qc, qp);
    gemm_bf16_kernel<<<gemm_grid(MROWS, QKDIM), 128, 0, stream>>>(
        xb, DIM, wkt, DIM, kb, QKDIM, MROWS, DIM);
    gemm_vt_kernel<<<gemm_grid(MROWS, HEADS * DVAL), 128, 0, stream>>>(xb, wvt, vt);
    gemm_relk_kernel<<<gemm_grid(NPOS, QKDIM), 128, 0, stream>>>(posb, wrelt, relk);

    // ---- attention, one (b,h) slice at a time (stream-serialized scratch reuse) ----
    for (int b = 0; b < BATCH; ++b) {
        for (int h = 0; h < HEADS; ++h) {
            const bf16* qcs = qc + (size_t)b * SEQ * QKDIM + h * DKEY;
            const bf16* qps = qp + (size_t)b * SEQ * QKDIM + h * DKEY;
            const bf16* kbs = kb + (size_t)b * SEQ * QKDIM + h * DKEY;
            const bf16* rks = relk + (size_t)h * NPOS * DKEY;
            const bf16* vts = vt + ((size_t)b * HEADS + h) * DVAL * SEQ;
            bf16* aos = ao + (size_t)b * SEQ * DIM + h * DVAL;
            gemm_f32_kernel<<<gemm_grid(SEQ, SEQ), 128, 0, stream>>>(
                qcs, QKDIM, kbs, QKDIM, logits, SEQ, SEQ, DKEY);
            gemm_f32_kernel<<<gemm_grid(SEQ, NPOS), 128, 0, stream>>>(
                qps, QKDIM, rks, DKEY, relbuf, NPOS, SEQ, DKEY);
            softmax_kernel<<<SEQ, 256, 0, stream>>>(logits, relbuf, attn);
            gemm_bf16_kernel<<<gemm_grid(SEQ, DVAL), 128, 0, stream>>>(
                attn, SEQ, vts, SEQ, aos, DIM, SEQ, SEQ);
        }
    }

    // ---- output projection ----
    gemm_final_kernel<<<gemm_grid(MROWS, DIM), 128, 0, stream>>>(ao, woutt, bout, out);
}
